// DecoderAttention_65824668778784
// MI455X (gfx1250) — matrix operand order repaired and checked by reading, not compiled, not run on hardware
//
#include <hip/hip_runtime.h>
#include <hip/hip_bf16.h>

// Problem constants (from reference)
#define V_  50257
#define E_  512
#define H_  1024
#define B_  64
#define S_  2048
#define KL_ 2560   // E + H (lstm_in) + H (s_t) concatenated K for gate GEMM
#define G4H 4096   // 4*H

typedef __attribute__((ext_vector_type(16))) __bf16 v16bf;
typedef __attribute__((ext_vector_type(8)))  __bf16 v8bf;
typedef __attribute__((ext_vector_type(8)))  float  v8f;
typedef __attribute__((ext_vector_type(2)))  float  v2f;

// Native CDNA5 transcendentals ---------------------------------------------
__device__ __forceinline__ float fast_tanh(float x) {
#if __has_builtin(__builtin_amdgcn_tanhf)
  return __builtin_amdgcn_tanhf(x);
#else
  float r;
  // V_TANH_F32 is a TRANS op: 1 independent op before the result is usable.
  asm("v_tanh_f32 %0, %1\n\tv_nop" : "=v"(r) : "v"(x));
  return r;
#endif
}
__device__ __forceinline__ float fast_sigmoid(float x) {
  // sigmoid(x) = 1 / (1 + 2^(-x*log2(e))) ; v_exp_f32 computes 2^x
  return __builtin_amdgcn_rcpf(1.0f + __builtin_amdgcn_exp2f(-1.442695040888963f * x));
}

// Pack 16 fp32 -> v16bf (A-operand fragment), RNE via v_cvt_pk_bf16_f32
__device__ __forceinline__ v16bf cvt16(float4 a0, float4 a1, float4 a2, float4 a3) {
  v16bf r;
  r[0]=(__bf16)a0.x;  r[1]=(__bf16)a0.y;  r[2]=(__bf16)a0.z;  r[3]=(__bf16)a0.w;
  r[4]=(__bf16)a1.x;  r[5]=(__bf16)a1.y;  r[6]=(__bf16)a1.z;  r[7]=(__bf16)a1.w;
  r[8]=(__bf16)a2.x;  r[9]=(__bf16)a2.y;  r[10]=(__bf16)a2.z; r[11]=(__bf16)a2.w;
  r[12]=(__bf16)a3.x; r[13]=(__bf16)a3.y; r[14]=(__bf16)a3.z; r[15]=(__bf16)a3.w;
  return r;
}

#define WMMA_BF16(A, B, C) \
  __builtin_amdgcn_wmma_f32_16x16x32_bf16(false, (A), false, (B), (short)0, (C), false, false)
#define WMMA_F32(A, B, C) \
  __builtin_amdgcn_wmma_f32_16x16x4_f32(false, (A), false, (B), (short)0, (C), false, false)

// ---------------------------------------------------------------------------
// fp32 -> bf16 bulk convert (8 elements/thread)
// ---------------------------------------------------------------------------
__global__ void cvt_bf16_kernel(const float* __restrict__ in, __bf16* __restrict__ out,
                                long long n8) {
  long long i = (long long)blockIdx.x * blockDim.x + threadIdx.x;
  if (i >= n8) return;
  const float4* p = (const float4*)in + i * 2;
  float4 a = p[0], c = p[1];
  v8bf o;
  o[0] = (__bf16)a.x; o[1] = (__bf16)a.y; o[2] = (__bf16)a.z; o[3] = (__bf16)a.w;
  o[4] = (__bf16)c.x; o[5] = (__bf16)c.y; o[6] = (__bf16)c.z; o[7] = (__bf16)c.w;
  *((v8bf*)out + i) = o;
}

// ---------------------------------------------------------------------------
// Wcat (4096 x 2560) bf16 = [Wih (4096x1536) | Whh (4096x1024)]
// ---------------------------------------------------------------------------
__global__ void wcat_kernel(const float* __restrict__ Wih, const float* __restrict__ Whh,
                            __bf16* __restrict__ Wcat) {
  int n = blockIdx.y;
  int k = blockIdx.x * 256 + threadIdx.x;
  float v = (k < 1536) ? Wih[(size_t)n * 1536 + k] : Whh[(size_t)n * 1024 + (k - 1536)];
  Wcat[(size_t)n * KL_ + k] = (__bf16)v;
}

// ---------------------------------------------------------------------------
// proj_s = s_t @ Wa_w^T + Wa_b (64x1024, tiny)
// ---------------------------------------------------------------------------
__global__ void projs_kernel(const float* __restrict__ hidden, const float* __restrict__ Wa_w,
                             const float* __restrict__ Wa_b, float* __restrict__ projs) {
  int idx = blockIdx.x * 256 + threadIdx.x;
  int b = idx >> 10, o = idx & (H_ - 1);
  const float4* hv = (const float4*)(hidden + (size_t)b * H_);
  const float4* wv = (const float4*)(Wa_w + (size_t)o * H_);
  float acc = 0.f;
  for (int k = 0; k < H_ / 4; ++k) {
    float4 a = hv[k], w = wv[k];
    acc += a.x * w.x + a.y * w.y + a.z * w.z + a.w * w.w;
  }
  projs[idx] = acc + Wa_b[o];
}

// ---------------------------------------------------------------------------
// Attention scores, fused: scores[b,s] = va . tanh(proj_s + Ua_b + enc@Ua^T) + va_b
// Wave tile = 32 s-rows x 64 o-cols: 2 A fragments x 4 B fragments -> 8 WMMA
// per K-step into 8 accumulators (1 KB of operands per WMMA from L0).
// B addresses are identical across the block's 4 waves -> WGP$ reuse.
// grid(S/128, B), block 128 (4 waves)
// ---------------------------------------------------------------------------
__global__ __launch_bounds__(128) void scores_wmma_kernel(
    const float* __restrict__ enc, const __bf16* __restrict__ uab,
    const float* __restrict__ projs, const float* __restrict__ ua_b,
    const float* __restrict__ va_w, const float* __restrict__ va_b,
    float* __restrict__ scores) {
  const int lane = threadIdx.x & 31;
  const int laneLo = lane & 15, laneHi = lane >> 4;
  const int b = blockIdx.y;
  const int s0 = (blockIdx.x * 4 + (threadIdx.x >> 5)) * 32;   // 32 s-rows per wave
  const float* Arow0 = enc + ((size_t)b * S_ + s0 + laneLo) * H_;
  const float* Arow1 = Arow0 + (size_t)16 * H_;

  float scAcc0[8], scAcc1[8];
#pragma unroll
  for (int r = 0; r < 8; ++r) { scAcc0[r] = 0.f; scAcc1[r] = 0.f; }

  for (int oG = 0; oG < H_ / 64; ++oG) {          // 16 groups of 4 o-tiles
    const int oBase = oG * 64;
    const __bf16* Bcol = uab + (size_t)(oBase + laneLo) * H_ + laneHi * 16;
    v8f acc0[4], acc1[4];
#pragma unroll
    for (int t = 0; t < 4; ++t) { acc0[t] = (v8f){}; acc1[t] = (v8f){}; }

#pragma unroll 2
    for (int k0 = 0; k0 < H_; k0 += 32) {
      const float4* a0lo = (const float4*)(Arow0 + k0 + laneHi * 16);
      const float4* a0hi = (const float4*)(Arow0 + k0 + 8 + laneHi * 16);
      v16bf a0 = cvt16(a0lo[0], a0lo[1], a0hi[0], a0hi[1]);
      const float4* a1lo = (const float4*)(Arow1 + k0 + laneHi * 16);
      const float4* a1hi = (const float4*)(Arow1 + k0 + 8 + laneHi * 16);
      v16bf a1 = cvt16(a1lo[0], a1lo[1], a1hi[0], a1hi[1]);
#pragma unroll
      for (int t = 0; t < 4; ++t) {
        v16bf bt = *(const v16bf*)(Bcol + (size_t)(16 * t) * H_ + k0);
        acc0[t] = WMMA_BF16(a0, bt, acc0[t]);
        acc1[t] = WMMA_BF16(a1, bt, acc1[t]);
      }
    }
#pragma unroll
    for (int t = 0; t < 4; ++t) {
      const int o = oBase + t * 16 + laneLo;
      const float ps = projs[(size_t)b * H_ + o] + ua_b[o];
      const float va = va_w[o];
#pragma unroll
      for (int r = 0; r < 8; ++r) {
        scAcc0[r] += va * fast_tanh(acc0[t][r] + ps);  // s = s0 + r + 8*laneHi
        scAcc1[r] += va * fast_tanh(acc1[t][r] + ps);  // s = s0 + 16 + r + 8*laneHi
      }
    }
  }
  // Sum across the 16 lanes of each half-wave (each half spans all o residues).
#pragma unroll
  for (int r = 0; r < 8; ++r) {
    float v0 = scAcc0[r], v1 = scAcc1[r];
    v0 += __shfl_xor(v0, 1, 32); v1 += __shfl_xor(v1, 1, 32);
    v0 += __shfl_xor(v0, 2, 32); v1 += __shfl_xor(v1, 2, 32);
    v0 += __shfl_xor(v0, 4, 32); v1 += __shfl_xor(v1, 4, 32);
    v0 += __shfl_xor(v0, 8, 32); v1 += __shfl_xor(v1, 8, 32);
    scAcc0[r] = v0; scAcc1[r] = v1;
  }
#pragma unroll
  for (int r = 0; r < 8; ++r)
    if (laneLo == r) {
      scores[(size_t)b * S_ + s0 + laneHi * 8 + r] = scAcc0[r] + va_b[0];
      scores[(size_t)b * S_ + s0 + 16 + laneHi * 8 + r] = scAcc1[r] + va_b[0];
    }
}

// ---------------------------------------------------------------------------
// Row softmax over S. grid(B), block 256.
// ---------------------------------------------------------------------------
__global__ void softmax_kernel(const float* __restrict__ scores, float* __restrict__ attn) {
  __shared__ float red[256];
  const int b = blockIdx.x, t = threadIdx.x;
  const float* row = scores + (size_t)b * S_;
  float mx = -3.0e38f;
  for (int s = t; s < S_; s += 256) mx = fmaxf(mx, row[s]);
  red[t] = mx; __syncthreads();
  for (int w = 128; w > 0; w >>= 1) {
    if (t < w) red[t] = fmaxf(red[t], red[t + w]);
    __syncthreads();
  }
  mx = red[0]; __syncthreads();
  float sum = 0.f;
  for (int s = t; s < S_; s += 256) {
    float e = __builtin_amdgcn_exp2f(1.442695040888963f * (row[s] - mx));
    attn[(size_t)b * S_ + s] = e;
    sum += e;
  }
  red[t] = sum; __syncthreads();
  for (int w = 128; w > 0; w >>= 1) {
    if (t < w) red[t] += red[t + w];
    __syncthreads();
  }
  float inv = 1.0f / red[0];
  for (int s = t; s < S_; s += 256) attn[(size_t)b * S_ + s] *= inv;
}

// ---------------------------------------------------------------------------
// context partials: ctxp[c][b][h] = sum_{s in chunk c} attn[b,s]*enc[b,s,h]
// grid(8, B), block 128 (each thread: 8 h via float4 pairs). Deterministic.
// ---------------------------------------------------------------------------
__global__ __launch_bounds__(128) void context_partial_kernel(
    const float* __restrict__ enc, const float* __restrict__ attn,
    float* __restrict__ ctxp) {
  const int b = blockIdx.y, chunk = blockIdx.x;
  const int h0 = threadIdx.x * 8;
  const float* base = enc + ((size_t)b * S_ + (size_t)chunk * 256) * H_ + h0;
  const float* arow = attn + (size_t)b * S_ + (size_t)chunk * 256;
  float acc[8];
#pragma unroll
  for (int i = 0; i < 8; ++i) acc[i] = 0.f;
  for (int s = 0; s < 256; ++s) {
    const float w = arow[s];
    const float4* p = (const float4*)(base + (size_t)s * H_);
    float4 x0 = p[0], x1 = p[1];
    acc[0] += w * x0.x; acc[1] += w * x0.y; acc[2] += w * x0.z; acc[3] += w * x0.w;
    acc[4] += w * x1.x; acc[5] += w * x1.y; acc[6] += w * x1.z; acc[7] += w * x1.w;
  }
  float* out = ctxp + ((size_t)chunk * B_ + b) * H_ + h0;
  ((float4*)out)[0] = make_float4(acc[0], acc[1], acc[2], acc[3]);
  ((float4*)out)[1] = make_float4(acc[4], acc[5], acc[6], acc[7]);
}

__global__ void context_reduce_kernel(const float* __restrict__ ctxp, float* __restrict__ ctx) {
  int idx = blockIdx.x * 256 + threadIdx.x;  // B*H
  float s = 0.f;
#pragma unroll
  for (int c = 0; c < 8; ++c) s += ctxp[(size_t)c * B_ * H_ + idx];
  ctx[idx] = s;
}

// ---------------------------------------------------------------------------
// lstm_in_ext bf16 (B x 2560) = [emb[x[b]] (512) | context (1024) | s_t (1024)]
// ---------------------------------------------------------------------------
__global__ void lstmin_kernel(const int* __restrict__ x, const float* __restrict__ emb,
                              const float* __restrict__ context, const float* __restrict__ hidden,
                              __bf16* __restrict__ lstmin) {
  const int b = blockIdx.y;
  const int k = blockIdx.x * 256 + threadIdx.x;
  float v;
  if (k < E_)            v = emb[(size_t)x[b] * E_ + k];
  else if (k < E_ + H_)  v = context[(size_t)b * H_ + (k - E_)];
  else                   v = hidden[(size_t)b * H_ + (k - E_ - H_)];
  lstmin[(size_t)b * KL_ + k] = (__bf16)v;
}

// ---------------------------------------------------------------------------
// gates (64 x 4096) = lstm_in_ext @ Wcat^T + bih + bhh, bf16 WMMA, K=2560.
// One wave -> one 16-col tile, ALL four 16-row M tiles (Wcat streamed once).
// grid(64), block 128 (4 waves -> 4 n-tiles each)
// ---------------------------------------------------------------------------
__global__ __launch_bounds__(128) void gates_wmma_kernel(
    const __bf16* __restrict__ lstmin, const __bf16* __restrict__ wcat,
    const float* __restrict__ bih, const float* __restrict__ bhh,
    float* __restrict__ gates) {
  const int lane = threadIdx.x & 31;
  const int laneLo = lane & 15, laneHi = lane >> 4;
  const int nt = blockIdx.x * 4 + (threadIdx.x >> 5);   // 0..255
  const int n = nt * 16 + laneLo;
  const __bf16* Bcol = wcat + (size_t)n * KL_ + laneHi * 16;
  const __bf16* A0 = lstmin + (size_t)laneLo * KL_;     // m-tile stride = 16*KL_
  v8f acc[4];
#pragma unroll
  for (int t = 0; t < 4; ++t) acc[t] = (v8f){};

#pragma unroll 2
  for (int k0 = 0; k0 < KL_; k0 += 32) {
    v16bf bm = *(const v16bf*)(Bcol + k0);
#pragma unroll
    for (int t = 0; t < 4; ++t) {
      const __bf16* Ar = A0 + (size_t)t * 16 * KL_;
      v8bf alo = *(const v8bf*)(Ar + k0 + laneHi * 16);
      v8bf ahi = *(const v8bf*)(Ar + k0 + 8 + laneHi * 16);
      v16bf a = __builtin_shufflevector(alo, ahi, 0,1,2,3,4,5,6,7,8,9,10,11,12,13,14,15);
      acc[t] = WMMA_BF16(a, bm, acc[t]);
    }
  }
  const float bias = bih[n] + bhh[n];
#pragma unroll
  for (int t = 0; t < 4; ++t)
#pragma unroll
    for (int r = 0; r < 8; ++r)
      gates[(size_t)(t * 16 + r + 8 * laneHi) * G4H + n] = acc[t][r] + bias;
}

// ---------------------------------------------------------------------------
// LSTM cell pointwise (native tanh/exp2). grid(256), block 256.
// ---------------------------------------------------------------------------
__global__ void lstm_cell_kernel(const float* __restrict__ gates, const float* __restrict__ cell,
                                 float* __restrict__ hnew, float* __restrict__ cnew) {
  int idx = blockIdx.x * 256 + threadIdx.x;
  int b = idx >> 10, h = idx & (H_ - 1);
  const float* g = gates + (size_t)b * G4H;
  float i_ = fast_sigmoid(g[h]);
  float f_ = fast_sigmoid(g[H_ + h]);
  float g_ = fast_tanh(g[2 * H_ + h]);
  float o_ = fast_sigmoid(g[3 * H_ + h]);
  float c = f_ * cell[idx] + i_ * g_;
  cnew[idx] = c;
  hnew[idx] = o_ * fast_tanh(c);
}

// ---------------------------------------------------------------------------
// predictions = h_new @ fc_w^T + fc_b via fp32 WMMA 16x16x4.
// One wave -> one 16-col tile, ALL four M tiles: fc_w (206 MB) streamed once
// (HBM-bound floor ~9 us; compute fits underneath).
// grid(1571), block 64 (2 waves; 3142 n-tiles)
// ---------------------------------------------------------------------------
__global__ __launch_bounds__(64) void fc_wmma_kernel(
    const float* __restrict__ hnew, const float* __restrict__ fcw,
    const float* __restrict__ fcb, float* __restrict__ pred) {
  const int lane = threadIdx.x & 31;
  const int laneLo = lane & 15, laneHi = lane >> 4;
  const int nt = blockIdx.x * 2 + (threadIdx.x >> 5);
  const int n = nt * 16 + laneLo;
  const int nr = (n < V_) ? n : (V_ - 1);
  const float* Bcol = fcw + (size_t)nr * H_ + 2 * laneHi;          // K = 2hi, 2hi+1
  const float* A0 = hnew + (size_t)laneLo * H_ + 2 * laneHi;       // m-tile stride 16*H_
  v8f acc[4];
#pragma unroll
  for (int t = 0; t < 4; ++t) acc[t] = (v8f){};

#pragma unroll 4
  for (int k0 = 0; k0 < H_; k0 += 4) {
    v2f bv = *(const v2f*)(Bcol + k0);
    acc[0] = WMMA_F32(*(const v2f*)(A0 + k0), bv, acc[0]);
    acc[1] = WMMA_F32(*(const v2f*)(A0 + (size_t)16 * H_ + k0), bv, acc[1]);
    acc[2] = WMMA_F32(*(const v2f*)(A0 + (size_t)32 * H_ + k0), bv, acc[2]);
    acc[3] = WMMA_F32(*(const v2f*)(A0 + (size_t)48 * H_ + k0), bv, acc[3]);
  }
  if (n < V_) {
    const float bias = fcb[n];
#pragma unroll
    for (int t = 0; t < 4; ++t)
#pragma unroll
      for (int r = 0; r < 8; ++r)
        pred[(size_t)(t * 16 + r + 8 * laneHi) * V_ + n] = acc[t][r] + bias;
  }
}

// ---------------------------------------------------------------------------
extern "C" void kernel_launch(void* const* d_in, const int* in_sizes, int n_in,
                              void* d_out, int out_size, void* d_ws, size_t ws_size,
                              hipStream_t stream) {
  const int*   x      = (const int*)d_in[0];
  const float* hidden = (const float*)d_in[1];
  const float* cell   = (const float*)d_in[2];
  const float* enc    = (const float*)d_in[3];
  const float* emb    = (const float*)d_in[4];
  const float* Wa_w   = (const float*)d_in[5];
  const float* Wa_b   = (const float*)d_in[6];
  const float* Ua_w   = (const float*)d_in[7];
  const float* Ua_b   = (const float*)d_in[8];
  const float* va_w   = (const float*)d_in[9];
  const float* va_b   = (const float*)d_in[10];
  const float* Wih    = (const float*)d_in[11];
  const float* Whh    = (const float*)d_in[12];
  const float* bih    = (const float*)d_in[13];
  const float* bhh    = (const float*)d_in[14];
  const float* fc_w   = (const float*)d_in[15];
  const float* fc_b   = (const float*)d_in[16];

  // Workspace layout (~27 MB)
  char* w = (char*)d_ws;
  __bf16* uab    = (__bf16*)w;  w += (size_t)H_ * H_ * 2;        // 2 MB
  __bf16* wcat   = (__bf16*)w;  w += (size_t)G4H * KL_ * 2;      // 20 MB
  float*  scores = (float*)w;   w += (size_t)B_ * S_ * 4;
  float*  attn   = (float*)w;   w += (size_t)B_ * S_ * 4;
  float*  projs  = (float*)w;   w += (size_t)B_ * H_ * 4;
  float*  ctx    = (float*)w;   w += (size_t)B_ * H_ * 4;
  float*  ctxp   = (float*)w;   w += (size_t)8 * B_ * H_ * 4;    // 2 MB
  __bf16* lstmin = (__bf16*)w;  w += (size_t)B_ * KL_ * 2;
  float*  gates  = (float*)w;   w += (size_t)B_ * G4H * 4;

  float* pred = (float*)d_out;
  float* hnew = pred + (size_t)B_ * V_;
  float* cnew = hnew + (size_t)B_ * H_;

  // Small weight conversions (enc stays fp32: converted in-register in scores)
  {
    long long n8 = (long long)H_ * H_ / 8;
    cvt_bf16_kernel<<<(unsigned)(n8 / 256), 256, 0, stream>>>(Ua_w, uab, n8);
  }
  wcat_kernel<<<dim3(KL_ / 256, G4H), 256, 0, stream>>>(Wih, Whh, wcat);

  projs_kernel<<<B_ * H_ / 256, 256, 0, stream>>>(hidden, Wa_w, Wa_b, projs);

  scores_wmma_kernel<<<dim3(S_ / 128, B_), 128, 0, stream>>>(enc, uab, projs, Ua_b,
                                                             va_w, va_b, scores);

  softmax_kernel<<<B_, 256, 0, stream>>>(scores, attn);

  context_partial_kernel<<<dim3(8, B_), 128, 0, stream>>>(enc, attn, ctxp);
  context_reduce_kernel<<<B_ * H_ / 256, 256, 0, stream>>>(ctxp, ctx);

  lstmin_kernel<<<dim3(KL_ / 256, B_), 256, 0, stream>>>(x, emb, ctx, hidden, lstmin);

  gates_wmma_kernel<<<64, 128, 0, stream>>>(lstmin, wcat, bih, bhh, gates);

  lstm_cell_kernel<<<B_ * H_ / 256, 256, 0, stream>>>(gates, cell, hnew, cnew);

  fc_wmma_kernel<<<1571, 64, 0, stream>>>(hnew, fc_w, fc_b, pred);

  (void)in_sizes; (void)n_in; (void)out_size; (void)ws_size;
}
